// PCT_23536420782438
// MI455X (gfx1250) — matrix operand order repaired and checked by reading, not compiled, not run on hardware
//
#include <hip/hip_runtime.h>
#include <hip/hip_bf16.h>
#include <stdint.h>

#define DEV static __device__ __forceinline__

typedef __attribute__((ext_vector_type(16))) __bf16 v16bf;
typedef __attribute__((ext_vector_type(8)))  float  v8f;

// ---------------- helpers ----------------
DEV uint16_t f2bf(float f) {
  union { float f; uint32_t u; } c; c.f = f;
  uint32_t u = c.u;
  u += 0x7FFFu + ((u >> 16) & 1u);   // round-to-nearest-even
  return (uint16_t)(u >> 16);
}
DEV float bf2f(uint16_t h) {
  union { uint32_t u; float f; } c; c.u = ((uint32_t)h) << 16;
  return c.f;
}
DEV float loadv(const float* p)    { return *p; }
DEV float loadv(const uint16_t* p) { return bf2f(*p); }
DEV void  storev(float* p, float v)    { *p = v; }
DEV void  storev(uint16_t* p, float v) { *p = f2bf(v); }

union FragU { v16bf v; uint4 q[2]; };

// A fragment (16x32 bf16): lane<16: row=r0+lane, K = k0+{0..7, 16..23}
//                          lane>=16: same rows,  K = k0+{8..15, 24..31}
DEV v16bf load_a_frag(const uint16_t* p, int ld, int row, int k0, int lane) {
  const int half = lane >> 4;
  const uint16_t* base = p + (size_t)row * ld + k0 + half * 8;
  FragU f;
  f.q[0] = *(const uint4*)(base);
  f.q[1] = *(const uint4*)(base + 16);
  return f.v;
}
// B fragment (32x16 bf16): lane l holds col l%16, K = k0 + (l<16 ? {0..7, 16..23} : {8..15, 24..31})
// "weights" stored row-major W[o, k] -> row index = output col.
DEV v16bf load_b_frag(const uint16_t* p, int ld, int row, int k0, int lane) {
  const int half = lane >> 4;
  const uint16_t* base = p + (size_t)row * ld + k0 + half * 8;
  FragU f;
  f.q[0] = *(const uint4*)(base);
  f.q[1] = *(const uint4*)(base + 16);
  return f.v;
}
DEV v8f wmma_bf(v16bf a, v16bf b, v8f c) {
  return __builtin_amdgcn_wmma_f32_16x16x32_bf16(false, a, false, b, (short)0, c, false, false);
}
DEV v8f zero8() {
  v8f z;
#pragma unroll
  for (int i = 0; i < 8; i++) z[i] = 0.f;
  return z;
}

// ---------------- generic bf16 WMMA GEMM ----------------
// Out[r, o] = sum_k A[r, k] * W[o, k]  (+ bias[o])
// A: [rows, K] bf16, W: [O, K] bf16. Block = 128 thr (4 waves), wave: 16 rows x 64 cols.
// OMODE: 0 = f32 out, 1 = bf16 out, 2 = f32 out + bf16 second output
template <int ACC_N, int OMODE, bool HAS_BIAS>
__global__ void gemm_kernel(const uint16_t* __restrict__ A, const uint16_t* __restrict__ W,
                            const float* __restrict__ bias, void* __restrict__ Out,
                            uint16_t* __restrict__ Out2, int K, int O) {
  const int lane = threadIdx.x & 31;
  const int wave = threadIdx.x >> 5;
  const int lr = lane & 15, half = lane >> 4;
  const int r0 = (blockIdx.x * (blockDim.x >> 5) + wave) * 16;
  const int c0 = blockIdx.y * 16 * ACC_N;

  v8f acc[ACC_N];
#pragma unroll
  for (int i = 0; i < ACC_N; i++) acc[i] = zero8();

  for (int k0 = 0; k0 < K; k0 += 32) {
    v16bf a = load_a_frag(A, K, r0 + lr, k0, lane);
#pragma unroll
    for (int i = 0; i < ACC_N; i++) {
      v16bf b = load_b_frag(W, K, c0 + i * 16 + lr, k0, lane);
      acc[i] = wmma_bf(a, b, acc[i]);
    }
  }
#pragma unroll
  for (int i = 0; i < ACC_N; i++) {
    const int cc = c0 + i * 16 + lr;
    const float bv = HAS_BIAS ? bias[cc] : 0.f;
#pragma unroll
    for (int j = 0; j < 8; j++) {
      const int m = half * 8 + j;
      const size_t idx = (size_t)(r0 + m) * O + cc;
      const float v = acc[i][j] + bv;
      if (OMODE == 1) ((uint16_t*)Out)[idx] = f2bf(v);
      else            ((float*)Out)[idx]    = v;
      if (OMODE == 2) Out2[idx] = f2bf(v);
    }
  }
}

// ---------------- attention pass 1: row softmax stats ----------------
// 1 wave per 32 queries (2 row tiles). Lane-local online softmax over the lane's
// private columns; single cross-lane (max,sum) combine at the end.
__global__ void attn_rowstats(const uint16_t* __restrict__ Qb, const uint16_t* __restrict__ KTb,
                              float* __restrict__ rowmax, float* __restrict__ rowsum,
                              int N, int M) {
  const int b = blockIdx.y;
  const int r0 = blockIdx.x * 32;
  const int lane = threadIdx.x & 31;
  const int lr = lane & 15, half = lane >> 4;
  const uint16_t* Q  = Qb  + (size_t)b * N * 64;
  const uint16_t* KT = KTb + (size_t)b * M * 64;

  v16bf qa[2][2];
#pragma unroll
  for (int rt = 0; rt < 2; rt++) {
    qa[rt][0] = load_a_frag(Q, 64, r0 + rt * 16 + lr, 0, lane);
    qa[rt][1] = load_a_frag(Q, 64, r0 + rt * 16 + lr, 32, lane);
  }
  float ml[2][8], sl[2][8];
#pragma unroll
  for (int rt = 0; rt < 2; rt++)
#pragma unroll
    for (int j = 0; j < 8; j++) { ml[rt][j] = -1e30f; sl[rt][j] = 0.f; }

  for (int mt = 0; mt < M; mt += 16) {
    if (mt + 16 + lr < M) __builtin_prefetch(KT + (size_t)(mt + 16 + lr) * 64, 0, 1);
    v16bf b0 = load_b_frag(KT, 64, mt + lr, 0, lane);
    v16bf b1 = load_b_frag(KT, 64, mt + lr, 32, lane);
#pragma unroll
    for (int rt = 0; rt < 2; rt++) {
      v8f s = wmma_bf(qa[rt][1], b1, wmma_bf(qa[rt][0], b0, zero8()));
#pragma unroll
      for (int j = 0; j < 8; j++) {
        const float v = s[j];
        const float nm = fmaxf(ml[rt][j], v);
        sl[rt][j] = sl[rt][j] * __expf(ml[rt][j] - nm) + __expf(v - nm);
        ml[rt][j] = nm;
      }
    }
  }
  // combine (max,sum) across the 16 lanes of each half
#pragma unroll
  for (int rt = 0; rt < 2; rt++) {
#pragma unroll
    for (int j = 0; j < 8; j++) {
      float m = ml[rt][j], s = sl[rt][j];
      for (int off = 1; off < 16; off <<= 1) {
        const float om = __shfl_xor(m, off, 32);
        const float os = __shfl_xor(s, off, 32);
        const float nm = fmaxf(m, om);
        s = s * __expf(m - nm) + os * __expf(om - nm);
        m = nm;
      }
      ml[rt][j] = m; sl[rt][j] = s;
    }
  }
  if (lr == 0) {
#pragma unroll
    for (int rt = 0; rt < 2; rt++)
#pragma unroll
      for (int j = 0; j < 8; j++) {
        const int row = r0 + rt * 16 + half * 8 + j;
        rowmax[(size_t)b * N + row] = ml[rt][j];
        rowsum[(size_t)b * N + row] = sl[rt][j];
      }
  }
}

// ---------------- attention pass 2: column sums of normalized softmax ----------------
__global__ void attn_colsum(const uint16_t* __restrict__ Qb, const uint16_t* __restrict__ KTb,
                            const float* __restrict__ rowmax, const float* __restrict__ rowsum,
                            float* __restrict__ colsum, int N, int M) {
  const int b = blockIdx.y;
  const int r0 = blockIdx.x * 32;
  const int lane = threadIdx.x & 31;
  const int lr = lane & 15, half = lane >> 4;
  const uint16_t* Q  = Qb  + (size_t)b * N * 64;
  const uint16_t* KT = KTb + (size_t)b * M * 64;

  v16bf qa[2][2];
  float rm[2][8], ri[2][8];
#pragma unroll
  for (int rt = 0; rt < 2; rt++) {
    qa[rt][0] = load_a_frag(Q, 64, r0 + rt * 16 + lr, 0, lane);
    qa[rt][1] = load_a_frag(Q, 64, r0 + rt * 16 + lr, 32, lane);
#pragma unroll
    for (int j = 0; j < 8; j++) {
      const int row = r0 + rt * 16 + half * 8 + j;
      rm[rt][j] = rowmax[(size_t)b * N + row];
      ri[rt][j] = 1.f / rowsum[(size_t)b * N + row];
    }
  }
  for (int mt = 0; mt < M; mt += 16) {
    if (mt + 16 + lr < M) __builtin_prefetch(KT + (size_t)(mt + 16 + lr) * 64, 0, 1);
    v16bf b0 = load_b_frag(KT, 64, mt + lr, 0, lane);
    v16bf b1 = load_b_frag(KT, 64, mt + lr, 32, lane);
    float cs = 0.f;
#pragma unroll
    for (int rt = 0; rt < 2; rt++) {
      v8f s = wmma_bf(qa[rt][1], b1, wmma_bf(qa[rt][0], b0, zero8()));
#pragma unroll
      for (int j = 0; j < 8; j++) cs += __expf(s[j] - rm[rt][j]) * ri[rt][j];
    }
    cs += __shfl_xor(cs, 16, 32);
    if (half == 0) atomicAdd(&colsum[(size_t)b * M + mt + lr], cs);
  }
}

// ---------------- v' = v / (1e-9 + colsum), stored transposed [B][64][M] ----------------
__global__ void scale_v(const float* __restrict__ v, const float* __restrict__ colsum,
                        uint16_t* __restrict__ vT, int mMask, int mShift, size_t n) {
  const size_t i = (size_t)blockIdx.x * blockDim.x + threadIdx.x;
  if (i >= n) return;
  const int d = (int)(i & 63);
  const size_t bm = i >> 6;
  const int m = (int)(bm & mMask);
  const int b = (int)(bm >> mShift);
  const float s = 1.f / (1e-9f + colsum[((size_t)b << mShift) + m]);
  vT[(((size_t)b * 64 + d) << mShift) + m] = f2bf(v[i] * s);
}

// ---------------- attention pass 3: x_r = P x v'^T, emit u = q - x_r (bf16) ----------------
__global__ void attn_av(const uint16_t* __restrict__ Qb, const uint16_t* __restrict__ KTb,
                        const uint16_t* __restrict__ VTb,
                        const float* __restrict__ rowmax, const float* __restrict__ rowsum,
                        const float* __restrict__ qf32, uint16_t* __restrict__ u_bf,
                        int N, int M) {
  __shared__ __align__(16) uint16_t Pl[16 * 32];
  const int b = blockIdx.y;
  const int r0 = blockIdx.x * 16;
  const int lane = threadIdx.x & 31;
  const int lr = lane & 15, half = lane >> 4;
  const uint16_t* Q  = Qb  + (size_t)b * N * 64;
  const uint16_t* KT = KTb + (size_t)b * M * 64;
  const uint16_t* VT = VTb + (size_t)b * 64 * M;

  const v16bf qa0 = load_a_frag(Q, 64, r0 + lr, 0, lane);
  const v16bf qa1 = load_a_frag(Q, 64, r0 + lr, 32, lane);

  float rm[8], ri[8];
#pragma unroll
  for (int j = 0; j < 8; j++) {
    const int row = r0 + half * 8 + j;
    rm[j] = rowmax[(size_t)b * N + row];
    ri[j] = 1.f / rowsum[(size_t)b * N + row];
  }
  v8f accx[4];
#pragma unroll
  for (int t = 0; t < 4; t++) accx[t] = zero8();

  for (int mc = 0; mc < M; mc += 32) {
    if (mc + 32 + lane < M) __builtin_prefetch(KT + (size_t)(mc + 32 + lane) * 64, 0, 1);
#pragma unroll
    for (int t = 0; t < 2; t++) {
      v16bf b0 = load_b_frag(KT, 64, mc + t * 16 + lr, 0, lane);
      v16bf b1 = load_b_frag(KT, 64, mc + t * 16 + lr, 32, lane);
      v8f s = wmma_bf(qa1, b1, wmma_bf(qa0, b0, zero8()));
#pragma unroll
      for (int j = 0; j < 8; j++)
        Pl[(half * 8 + j) * 32 + t * 16 + lr] = f2bf(__expf(s[j] - rm[j]) * ri[j]);
    }
    __syncthreads();
    v16bf a = load_a_frag((const uint16_t*)Pl, 32, lr, 0, lane);
#pragma unroll
    for (int t = 0; t < 4; t++) {
      v16bf bm = load_b_frag(VT, M, t * 16 + lr, mc, lane);
      accx[t] = wmma_bf(a, bm, accx[t]);
    }
    __syncthreads();
  }
#pragma unroll
  for (int t = 0; t < 4; t++) {
#pragma unroll
    for (int j = 0; j < 8; j++) {
      const int row = r0 + half * 8 + j;
      const int d = t * 16 + lr;
      const size_t idx = ((size_t)b * N + row) * 64 + d;
      u_bf[idx] = f2bf(qf32[idx] - accx[t][j]);
    }
  }
}

// ---------------- GroupNorm stats ----------------
template <typename T>
__global__ void gn_stats(const T* __restrict__ x, float* __restrict__ mean, float* __restrict__ rstd,
                         int L, int C, int G, int cg, int cgShift, float eps) {
  const int bg = blockIdx.x;
  const int b = bg / G, g = bg % G;
  const size_t base = (size_t)b * L * C;
  const int n = L * cg;
  float s = 0.f, ss = 0.f;
  for (int i = threadIdx.x; i < n; i += blockDim.x) {
    const int l = i >> cgShift, c = g * cg + (i & (cg - 1));
    const float v = loadv(&x[base + (size_t)l * C + c]);
    s += v; ss += v * v;
  }
  __shared__ float sh0[256], sh1[256];
  sh0[threadIdx.x] = s; sh1[threadIdx.x] = ss;
  __syncthreads();
  for (int off = 128; off > 0; off >>= 1) {
    if ((int)threadIdx.x < off) { sh0[threadIdx.x] += sh0[threadIdx.x + off]; sh1[threadIdx.x] += sh1[threadIdx.x + off]; }
    __syncthreads();
  }
  if (threadIdx.x == 0) {
    const float mu = sh0[0] / n;
    const float var = sh1[0] / n - mu * mu;
    mean[bg] = mu;
    rstd[bg] = rsqrtf(var + eps);
  }
}

// ---------------- GroupNorm apply (+ act, + optional residual) ----------------
// ACT: 0=relu, 1=leaky(0.2). All of C, L, C/G are powers of two -> shift/mask indexing.
template <typename InT, typename OutT, int ACT, bool ADD_RES>
__global__ void gn_apply(const InT* __restrict__ x, const float* __restrict__ res,
                         const float* __restrict__ w, const float* __restrict__ bgn,
                         const float* __restrict__ mean, const float* __restrict__ rstd,
                         OutT* __restrict__ dst, size_t n, int cShift, int lShift, int cgShift,
                         int G, int dstStride, int dstOff) {
  const size_t i = (size_t)blockIdx.x * blockDim.x + threadIdx.x;
  if (i >= n) return;
  const int c = (int)(i & ((1u << cShift) - 1u));
  const size_t bl = i >> cShift;
  const int b = (int)(bl >> lShift);
  const int g = c >> cgShift;
  float v = loadv(&x[i]);
  v = (v - mean[b * G + g]) * rstd[b * G + g] * w[c] + bgn[c];
  v = (ACT == 0) ? fmaxf(v, 0.f) : (v > 0.f ? v : 0.2f * v);
  if (ADD_RES) v += res[i];
  storev(&dst[bl * dstStride + dstOff + c], v);
}

// ---------------- misc elementwise ----------------
__global__ void zero_u16(uint16_t* p, size_t n) {
  const size_t i = (size_t)blockIdx.x * blockDim.x + threadIdx.x;
  if (i < n) p[i] = 0;
}
__global__ void zero_f32(float* p, size_t n) {
  const size_t i = (size_t)blockIdx.x * blockDim.x + threadIdx.x;
  if (i < n) p[i] = 0.f;
}
// src [B, C, L] f32 -> dst[(b*L+l)*stride + off + c] bf16
__global__ void tpose_bf(const float* __restrict__ src, uint16_t* __restrict__ dst,
                         int C, int L, int stride, int off, size_t n) {
  const size_t i = (size_t)blockIdx.x * blockDim.x + threadIdx.x;
  if (i >= n) return;
  const int l = (int)(i % L);
  const size_t r = i / L;
  const int c = (int)(r % C);
  const int b = (int)(r / C);
  dst[((size_t)b * L + l) * stride + off + c] = f2bf(src[i]);
}
// src [O, C] f32 -> dst[o*stride + off + c] bf16
__global__ void pack_w(const float* __restrict__ src, uint16_t* __restrict__ dst,
                       int O, int C, int stride, int off) {
  const int i = blockIdx.x * blockDim.x + threadIdx.x;
  if (i >= O * C) return;
  const int c = i % C, o = i / C;
  dst[(size_t)o * stride + off + c] = f2bf(src[i]);
}

// ---------------- host orchestration ----------------
extern "C" void kernel_launch(void* const* d_in, const int* in_sizes, int n_in,
                              void* d_out, int out_size, void* d_ws, size_t ws_size,
                              hipStream_t stream) {
  (void)in_sizes; (void)n_in; (void)out_size; (void)ws_size;
  const int B = 16, N = 4096, M = 1024;
  const float eps = 1e-5f;

  const float* x        = (const float*)d_in[0];
  const float* x_global = (const float*)d_in[1];
  const float* xyz      = (const float*)d_in[2];
  const float* xyzg     = (const float*)d_in[3];
  const float* conv1_w  = (const float*)d_in[4];
  const float* gn1_w    = (const float*)d_in[5];
  const float* gn1_b    = (const float*)d_in[6];
  const float* conv2_w  = (const float*)d_in[7];
  const float* gn2_w    = (const float*)d_in[8];
  const float* gn2_b    = (const float*)d_in[9];
  const float* qk_w     = (const float*)d_in[10];
  const float* v_w      = (const float*)d_in[11];
  const float* v_b      = (const float*)d_in[12];
  const float* t_w      = (const float*)d_in[13];
  const float* t_b      = (const float*)d_in[14];
  const float* sgn_w    = (const float*)d_in[15];
  const float* sgn_b    = (const float*)d_in[16];
  const float* p_w      = (const float*)d_in[17];
  const float* pg_w     = (const float*)d_in[18];
  const float* fuse_w   = (const float*)d_in[19];
  const float* fgn_w    = (const float*)d_in[20];
  const float* fgn_b    = (const float*)d_in[21];
  const float* c1_w     = (const float*)d_in[22];
  const float* c1_b     = (const float*)d_in[23];
  const float* g1_w     = (const float*)d_in[24];
  const float* g1_b     = (const float*)d_in[25];
  const float* c2_w     = (const float*)d_in[26];
  const float* c2_b     = (const float*)d_in[27];
  const float* g2_w     = (const float*)d_in[28];
  const float* g2_b     = (const float*)d_in[29];

  uint8_t* w8 = (uint8_t*)d_ws;
  size_t off = 0;
  auto alloc = [&](size_t bytes) -> void* {
    off = (off + 255) & ~(size_t)255;
    void* p = w8 + off;
    off += bytes;
    return p;
  };
  uint16_t* Xin  = (uint16_t*)alloc((size_t)B * N * 32 * 2);
  uint16_t* Xg   = (uint16_t*)alloc((size_t)B * M * 32 * 2);
  uint16_t* Xq   = (uint16_t*)alloc((size_t)B * N * 96 * 2);
  uint16_t* Xkv  = (uint16_t*)alloc((size_t)B * M * 96 * 2);
  uint16_t* outs = (uint16_t*)alloc((size_t)B * N * 256 * 2);
  float*    qf   = (float*)   alloc((size_t)B * N * 64 * 4);
  uint16_t* qbf  = (uint16_t*)alloc((size_t)B * N * 64 * 2);
  uint16_t* kT   = (uint16_t*)alloc((size_t)B * M * 64 * 2);
  float*    vf   = (float*)   alloc((size_t)B * M * 64 * 4);
  uint16_t* vT   = (uint16_t*)alloc((size_t)B * 64 * M * 2);
  float*    rmax = (float*)   alloc((size_t)B * N * 4);
  float*    rsum = (float*)   alloc((size_t)B * N * 4);
  float*    csum = (float*)   alloc((size_t)B * M * 4);
  uint16_t* ubf  = (uint16_t*)alloc((size_t)B * N * 64 * 2);
  float*    tpre = (float*)   alloc((size_t)B * N * 64 * 4);   // also conv1 pre-act
  float*    hgp  = (float*)   alloc((size_t)B * M * 64 * 4);
  float*    xfp  = (float*)   alloc((size_t)B * N * 128 * 4);
  uint16_t* xfbf = (uint16_t*)alloc((size_t)B * N * 128 * 2);
  uint16_t* s1bf = (uint16_t*)alloc((size_t)B * N * 512 * 2);
  float*    mean = (float*)   alloc((size_t)B * 16 * 4);
  float*    rstd = (float*)   alloc((size_t)B * 16 * 4);
  uint16_t* Wc1  = (uint16_t*)alloc(64 * 32 * 2);
  uint16_t* Wc2  = (uint16_t*)alloc(64 * 32 * 2);
  uint16_t* Wq   = (uint16_t*)alloc(64 * 96 * 2);
  uint16_t* Wk   = (uint16_t*)alloc(64 * 96 * 2);
  uint16_t* Wv   = (uint16_t*)alloc(64 * 96 * 2);
  uint16_t* Wt   = (uint16_t*)alloc(64 * 64 * 2);
  uint16_t* Wf   = (uint16_t*)alloc(128 * 256 * 2);
  uint16_t* Ws1  = (uint16_t*)alloc(512 * 128 * 2);
  uint16_t* Ws2  = (uint16_t*)alloc(128 * 512 * 2);

  auto g1 = [](size_t n) { return dim3((unsigned)((n + 255) / 256)); };

  // ---- one-time zero of padded regions ----
  zero_u16<<<g1((size_t)B * N * 32), 256, 0, stream>>>(Xin, (size_t)B * N * 32);
  zero_u16<<<g1((size_t)B * N * 96), 256, 0, stream>>>(Xq,  (size_t)B * N * 96);
  zero_u16<<<g1((size_t)B * M * 96), 256, 0, stream>>>(Xkv, (size_t)B * M * 96);
  zero_u16<<<g1(64 * 32), 256, 0, stream>>>(Wc1, 64 * 32);
  zero_u16<<<g1(64 * 96), 256, 0, stream>>>(Wq, 64 * 96);
  zero_u16<<<g1(64 * 96), 256, 0, stream>>>(Wk, 64 * 96);
  zero_u16<<<g1(64 * 96), 256, 0, stream>>>(Wv, 64 * 96);

  // ---- transpose inputs / pack static weights ----
  tpose_bf<<<g1((size_t)B * 6 * N), 256, 0, stream>>>(x, Xin, 6, N, 32, 0, (size_t)B * 6 * N);
  tpose_bf<<<g1((size_t)B * 32 * M), 256, 0, stream>>>(x_global, Xg, 32, M, 32, 0, (size_t)B * 32 * M);
  tpose_bf<<<g1((size_t)B * 3 * N), 256, 0, stream>>>(xyz, Xq, 3, N, 96, 64, (size_t)B * 3 * N);
  tpose_bf<<<g1((size_t)B * 3 * M), 256, 0, stream>>>(xyzg, Xkv, 3, M, 96, 64, (size_t)B * 3 * M);
  pack_w<<<g1(64 * 6), 256, 0, stream>>>(conv1_w, Wc1, 64, 6, 32, 0);
  pack_w<<<g1(64 * 32), 256, 0, stream>>>(conv2_w, Wc2, 64, 32, 32, 0);
  pack_w<<<g1(128 * 256), 256, 0, stream>>>(fuse_w, Wf, 128, 256, 256, 0);
  pack_w<<<g1(512 * 128), 256, 0, stream>>>(c1_w, Ws1, 512, 128, 128, 0);
  pack_w<<<g1(128 * 512), 256, 0, stream>>>(c2_w, Ws2, 128, 512, 512, 0);

  const size_t rowsN = (size_t)B * N, rowsM = (size_t)B * M;

  // ---- h = relu(gn(conv1(x))) -> Xq[:, 0:64] ----
  gemm_kernel<4, 0, false><<<dim3((unsigned)(rowsN / 64), 1), 128, 0, stream>>>(Xin, Wc1, nullptr, tpre, nullptr, 32, 64);
  gn_stats<float><<<B * 16, 256, 0, stream>>>(tpre, mean, rstd, N, 64, 16, 4, 2, eps);
  gn_apply<float, uint16_t, 0, false><<<g1(rowsN * 64), 256, 0, stream>>>(
      tpre, nullptr, gn1_w, gn1_b, mean, rstd, Xq, rowsN * 64, 6, 12, 2, 16, 96, 0);

  // ---- hg = relu(gn(conv2(x_global))) -> Xkv[:, 0:64] ----
  gemm_kernel<4, 0, false><<<dim3((unsigned)(rowsM / 64), 1), 128, 0, stream>>>(Xg, Wc2, nullptr, hgp, nullptr, 32, 64);
  gn_stats<float><<<B * 16, 256, 0, stream>>>(hgp, mean, rstd, M, 64, 16, 4, 2, eps);
  gn_apply<float, uint16_t, 0, false><<<g1(rowsM * 64), 256, 0, stream>>>(
      hgp, nullptr, gn2_w, gn2_b, mean, rstd, Xkv, rowsM * 64, 6, 10, 2, 16, 96, 0);

  // ---- 4 SA layers ----
  for (int i = 0; i < 4; i++) {
    pack_w<<<g1(64 * 64), 256, 0, stream>>>(qk_w + i * 4096, Wq, 64, 64, 96, 0);
    pack_w<<<g1(64 * 3), 256, 0, stream>>>(p_w + i * 192, Wq, 64, 3, 96, 64);
    pack_w<<<g1(64 * 64), 256, 0, stream>>>(qk_w + i * 4096, Wk, 64, 64, 96, 0);
    pack_w<<<g1(64 * 3), 256, 0, stream>>>(pg_w + i * 192, Wk, 64, 3, 96, 64);
    pack_w<<<g1(64 * 64), 256, 0, stream>>>(v_w + i * 4096, Wv, 64, 64, 96, 0);
    pack_w<<<g1(64 * 3), 256, 0, stream>>>(pg_w + i * 192, Wv, 64, 3, 96, 64);
    pack_w<<<g1(64 * 64), 256, 0, stream>>>(t_w + i * 4096, Wt, 64, 64, 64, 0);

    // q (f32 + bf16 fused epilogue), k (bf16), v (f32, +bias)
    gemm_kernel<4, 2, false><<<dim3((unsigned)(rowsN / 64), 1), 128, 0, stream>>>(Xq, Wq, nullptr, qf, qbf, 96, 64);
    gemm_kernel<4, 1, false><<<dim3((unsigned)(rowsM / 64), 1), 128, 0, stream>>>(Xkv, Wk, nullptr, kT, nullptr, 96, 64);
    gemm_kernel<4, 0, true><<<dim3((unsigned)(rowsM / 64), 1), 128, 0, stream>>>(Xkv, Wv, v_b + i * 64, vf, nullptr, 96, 64);

    // attention (3 sweeps, never materialize B x N x M)
    attn_rowstats<<<dim3(N / 32, B), 32, 0, stream>>>(qbf, kT, rmax, rsum, N, M);
    zero_f32<<<g1((size_t)B * M), 256, 0, stream>>>(csum, (size_t)B * M);
    attn_colsum<<<dim3(N / 32, B), 32, 0, stream>>>(qbf, kT, rmax, rsum, csum, N, M);
    scale_v<<<g1(rowsM * 64), 256, 0, stream>>>(vf, csum, vT, M - 1, 10, rowsM * 64);
    attn_av<<<dim3(N / 16, B), 32, 0, stream>>>(qbf, kT, vT, rmax, rsum, qf, ubf, N, M);

    // t = relu(gn(trans(q - x_r))) ; cur = q + t
    gemm_kernel<4, 0, true><<<dim3((unsigned)(rowsN / 64), 1), 128, 0, stream>>>(ubf, Wt, t_b + i * 64, tpre, nullptr, 64, 64);
    gn_stats<float><<<B * 4, 256, 0, stream>>>(tpre, mean, rstd, N, 64, 4, 16, 4, eps);
    gn_apply<float, uint16_t, 0, true><<<g1(rowsN * 64), 256, 0, stream>>>(
        tpre, qf, sgn_w + i * 64, sgn_b + i * 64, mean, rstd, Xq, rowsN * 64, 6, 12, 4, 4, 96, 0);
    gn_apply<float, uint16_t, 0, true><<<g1(rowsN * 64), 256, 0, stream>>>(
        tpre, qf, sgn_w + i * 64, sgn_b + i * 64, mean, rstd, outs, rowsN * 64, 6, 12, 4, 4, 256, 64 * i);
  }

  // ---- fuse: leaky(gn(fuse_w @ concat)) ----
  gemm_kernel<4, 0, false><<<dim3((unsigned)(rowsN / 64), 2), 128, 0, stream>>>(outs, Wf, nullptr, xfp, nullptr, 256, 128);
  gn_stats<float><<<B * 16, 256, 0, stream>>>(xfp, mean, rstd, N, 128, 16, 8, 3, eps);
  gn_apply<float, uint16_t, 1, false><<<g1(rowsN * 128), 256, 0, stream>>>(
      xfp, nullptr, fgn_w, fgn_b, mean, rstd, xfbf, rowsN * 128, 7, 12, 3, 16, 128, 0);

  // ---- convs1: relu(gn(...)) ----
  gemm_kernel<4, 1, true><<<dim3((unsigned)(rowsN / 64), 8), 128, 0, stream>>>(xfbf, Ws1, c1_b, s1bf, nullptr, 128, 512);
  gn_stats<uint16_t><<<B * 8, 256, 0, stream>>>(s1bf, mean, rstd, N, 512, 8, 64, 6, eps);
  gn_apply<uint16_t, uint16_t, 0, false><<<g1(rowsN * 512), 256, 0, stream>>>(
      s1bf, nullptr, g1_w, g1_b, mean, rstd, s1bf, rowsN * 512, 9, 12, 6, 8, 512, 0);

  // ---- convs2: relu(gn(...)) -> d_out [B, N, 128] ----
  float* outp = (float*)d_out;
  gemm_kernel<4, 0, true><<<dim3((unsigned)(rowsN / 64), 2), 128, 0, stream>>>(s1bf, Ws2, c2_b, outp, nullptr, 512, 128);
  gn_stats<float><<<B * 4, 256, 0, stream>>>(outp, mean, rstd, N, 128, 4, 32, 5, eps);
  gn_apply<float, float, 0, false><<<g1(rowsN * 128), 256, 0, stream>>>(
      outp, nullptr, g2_w, g2_b, mean, rstd, outp, rowsN * 128, 7, 12, 5, 4, 128, 0);
}
